// TMDIntegrationLayer_41712722379175
// MI455X (gfx1250) — compile-verified, matrix operand order checked by reading, not run on hardware
//
#include <hip/hip_runtime.h>
#include <hip/hip_bf16.h>
#include <math.h>

// ---------------------------------------------------------------------------
// TMD integration layer for gfx1250 (MI455X, wave32, WMMA).
//
// One wave32 handles one batch element b:
//   nn1 : 1 tile of 16 rows  (rows = k bins, 10 valid)
//   nn2 : 7 tiles of 16 rows (rows = flattened (k,phi), 100 valid)
// Each tile: layer0 on VALU directly in WMMA A-layout, layers 1&2 as
// v_wmma_f32_16x16x32_f16 (2 per layer for the 32 output cols), one
// D->A transpose through LDS, layer3 as per-lane dot + shfl_xor butterfly.
// ---------------------------------------------------------------------------

typedef __attribute__((ext_vector_type(16))) _Float16 v16h;
typedef __attribute__((ext_vector_type(8)))  _Float16 v8h;
typedef __attribute__((ext_vector_type(8)))  float    v8f;

#define BTOT     32768
#define KBINS    10
#define PBINS    10
#define NWAVES   8          // waves per block
#define NTHREADS (NWAVES * 32)

// per-net LDS layout (floats): W0 @0 (2x32 row-major), b0 @64, W1 @96 (32x32),
// b1 @1120, W2 @1152, b2 @2176, W3 @2208, b3 @2240.  net stride 2304.
#define NETSTRIDE 2304

__device__ __forceinline__ float relu6f(float x) { return fminf(fmaxf(x, 0.0f), 6.0f); }

// ISA 16-bit A/B operand element->K mapping (CDNA5 ISA 7.12.2):
// lanes 0-15: e=0..7 -> K=e, e=8..15 -> K=16+(e-8); lanes 16-31: +8.
__device__ __forceinline__ int kmap(int e, int hi) {
    return ((e & 8) << 1) + (e & 7) + (hi << 3);
}

struct NetConsts {
    float c0[16], w1r[16];                 // layer0: h = relu6(c0 + feat*w1r)
    v16h  B1lo, B1hi, B2lo, B2hi;          // W1/W2 in WMMA B layout (f16)
    float b1lo, b1hi, b2lo, b2hi;          // bias per owned column
    float w3a, w3b, b3v;                   // layer3 weights for owned columns
};

__device__ __forceinline__ void loadNet(NetConsts& nc, const float* w, float a,
                                        int nl, int hi) {
#pragma unroll
    for (int e = 0; e < 16; ++e) {
        const int j = kmap(e, hi);
        nc.c0[e]  = fmaf(a, w[j], w[64 + j]);       // a*W0[0][j] + b0[j]
        nc.w1r[e] = w[32 + j];                      // W0[1][j]
        nc.B1lo[e] = (_Float16)w[96   + j * 32 + nl];
        nc.B1hi[e] = (_Float16)w[96   + j * 32 + 16 + nl];
        nc.B2lo[e] = (_Float16)w[1152 + j * 32 + nl];
        nc.B2hi[e] = (_Float16)w[1152 + j * 32 + 16 + nl];
    }
    nc.b1lo = w[1120 + nl];      nc.b1hi = w[1120 + 16 + nl];
    nc.b2lo = w[2176 + nl];      nc.b2hi = w[2176 + 16 + nl];
    nc.w3a  = w[2208 + nl];      nc.w3b  = w[2208 + 16 + nl];
    nc.b3v  = w[2240];
}

// Run the 2->32->32->32->1 MLP on a 16-row tile. feat = row feature for
// row m = (lane&15). Result: fin[v] = relu6 MLP output of row (v + 8*(lane>>4)),
// replicated across all 16 lanes of the half.
__device__ __forceinline__ void mlpTile(const NetConsts& nc, float feat,
                                        _Float16* hb, int nl, int hi,
                                        float fin[8]) {
    // ---- layer 0 (VALU, directly in A layout) ----
    v16h a0;
#pragma unroll
    for (int e = 0; e < 16; ++e)
        a0[e] = (_Float16)relu6f(fmaf(feat, nc.w1r[e], nc.c0[e]));

    // ---- layer 1: two 16x16x32 WMMAs, bias in C ----
    v8f c;
#pragma unroll
    for (int v = 0; v < 8; ++v) c[v] = nc.b1lo;
    v8f dlo = __builtin_amdgcn_wmma_f32_16x16x32_f16(false, a0, false, nc.B1lo,
                                                     (short)0, c, false, false);
#pragma unroll
    for (int v = 0; v < 8; ++v) c[v] = nc.b1hi;
    v8f dhi = __builtin_amdgcn_wmma_f32_16x16x32_f16(false, a0, false, nc.B1hi,
                                                     (short)0, c, false, false);

    // ---- relu6 + D-layout -> A-layout transpose through LDS ----
    // D layout: lane owns column nl (and 16+nl), rows v + 8*hi.
#pragma unroll
    for (int v = 0; v < 8; ++v) {
        const int m = v + (hi << 3);
        hb[m * 32 + nl]      = (_Float16)relu6f(dlo[v]);
        hb[m * 32 + 16 + nl] = (_Float16)relu6f(dhi[v]);
    }
    asm volatile("s_wait_dscnt 0" ::: "memory");
    // A layout needs row m=nl, K chunks [8*hi..8*hi+7] and [16+8*hi..+7]:
    // both contiguous 16B -> ds_load_b128.
    const v8h p0 = *(const v8h*)(hb + nl * 32 + (hi << 3));
    const v8h p1 = *(const v8h*)(hb + nl * 32 + 16 + (hi << 3));
    v16h a1;
#pragma unroll
    for (int e = 0; e < 8; ++e) { a1[e] = p0[e]; a1[8 + e] = p1[e]; }

    // ---- layer 2: two more WMMAs ----
#pragma unroll
    for (int v = 0; v < 8; ++v) c[v] = nc.b2lo;
    v8f elo = __builtin_amdgcn_wmma_f32_16x16x32_f16(false, a1, false, nc.B2lo,
                                                     (short)0, c, false, false);
#pragma unroll
    for (int v = 0; v < 8; ++v) c[v] = nc.b2hi;
    v8f ehi = __builtin_amdgcn_wmma_f32_16x16x32_f16(false, a1, false, nc.B2hi,
                                                     (short)0, c, false, false);

    // ---- layer 3: per-lane dot over owned columns + butterfly over 16 lanes ----
#pragma unroll
    for (int v = 0; v < 8; ++v) {
        float p = fmaf(relu6f(elo[v]), nc.w3a, relu6f(ehi[v]) * nc.w3b);
        p += __shfl_xor(p, 1, 32);
        p += __shfl_xor(p, 2, 32);
        p += __shfl_xor(p, 4, 32);
        p += __shfl_xor(p, 8, 32);
        fin[v] = relu6f(p + nc.b3v);
    }
}

__global__ __launch_bounds__(NTHREADS)
void tmd_integration_kernel(const float* __restrict__ qT,
                            const float* __restrict__ x1,
                            const float* __restrict__ x2,
                            const float* n1W0, const float* n1b0,
                            const float* n1W1, const float* n1b1,
                            const float* n1W2, const float* n1b2,
                            const float* n1W3, const float* n1b3,
                            const float* n2W0, const float* n2b0,
                            const float* n2W1, const float* n2b1,
                            const float* n2W2, const float* n2b2,
                            const float* n2W3, const float* n2b3,
                            float* __restrict__ out) {
    __shared__ float sW[2 * NETSTRIDE];                          // staged weights
    __shared__ __attribute__((aligned(16))) _Float16 sH[NWAVES][16 * 32];
    __shared__ float sWk[NWAVES][16];                            // nn1[k]*k_vals[k]

    const int tid = threadIdx.x;

    // ---- cooperative weight staging into LDS ----
    {
        const float* srcs[16] = {n1W0, n1b0, n1W1, n1b1, n1W2, n1b2, n1W3, n1b3,
                                 n2W0, n2b0, n2W1, n2b1, n2W2, n2b2, n2W3, n2b3};
        const int offs[16] = {0, 64, 96, 1120, 1152, 2176, 2208, 2240,
                              NETSTRIDE + 0,    NETSTRIDE + 64,   NETSTRIDE + 96,
                              NETSTRIDE + 1120, NETSTRIDE + 1152, NETSTRIDE + 2176,
                              NETSTRIDE + 2208, NETSTRIDE + 2240};
        const int lens[16] = {64, 32, 1024, 32, 1024, 32, 32, 1,
                              64, 32, 1024, 32, 1024, 32, 32, 1};
#pragma unroll
        for (int i = 0; i < 16; ++i) {
            const float* s = srcs[i];
            float* d = sW + offs[i];
            for (int j = tid; j < lens[i]; j += NTHREADS) d[j] = s[j];
        }
    }
    __syncthreads();

    const int wave = tid >> 5;
    const int lane = tid & 31;
    const int nl = lane & 15;
    const int hi = lane >> 4;
    const int b  = blockIdx.x * NWAVES + wave;

    const float dk   = 10.0f / 9.0f;
    const float dphi = 3.14159265358979323846f / 9.0f;
    const float q  = qT[b];
    const float a1 = x1[b];
    const float a2 = x2[b];

    _Float16* hb = &sH[wave][0];
    float*    wk = &sWk[wave][0];

    float fin[8];
    NetConsts nc;

    // ================= nn1: one tile, rows m = k-bin ==================
    loadNet(nc, sW, a1, nl, hi);
    {
        const float kv = (float)nl * dk;       // feature for row m = nl
        mlpTile(nc, kv, hb, nl, hi, fin);
        if (nl == 0) {
#pragma unroll
            for (int v = 0; v < 8; ++v) {
                const int m = v + (hi << 3);
                if (m < KBINS) wk[m] = fin[v] * ((float)m * dk);
            }
        }
    }
    asm volatile("s_wait_dscnt 0" ::: "memory");

    // ================= nn2: 7 tiles over the (k,phi) grid =============
    loadNet(nc, sW + NETSTRIDE, a2, nl, hi);
    const float q2 = q * q;
    float acc = 0.0f;
    for (int t = 0; t < 7; ++t) {
        const int   r   = t * 16 + nl;
        const int   k   = r / 10;
        const int   ph  = r - k * 10;
        const float kv  = (float)k * dk;
        const float cph = __cosf((float)ph * dphi);
        const float kb  = sqrtf(fmaxf(q2 + kv * kv - 2.0f * q * kv * cph, 0.0f));
        mlpTile(nc, kb, hb, nl, hi, fin);
        if (nl == 0) {
#pragma unroll
            for (int v = 0; v < 8; ++v) {
                const int rr = t * 16 + v + (hi << 3);
                if (rr < KBINS * PBINS) acc += fin[v] * wk[rr / 10];
            }
        }
    }

    // partials live on lanes 0 and 16 only
    acc += __shfl_xor(acc, 16, 32);
    if (lane == 0) out[b] = acc * dk * dphi;
}

extern "C" void kernel_launch(void* const* d_in, const int* in_sizes, int n_in,
                              void* d_out, int out_size, void* d_ws, size_t ws_size,
                              hipStream_t stream) {
    (void)in_sizes; (void)n_in; (void)out_size; (void)d_ws; (void)ws_size;
    const float* qT = (const float*)d_in[0];
    const float* x1 = (const float*)d_in[1];
    const float* x2 = (const float*)d_in[2];
    const float* n1W0 = (const float*)d_in[3];
    const float* n1b0 = (const float*)d_in[4];
    const float* n1W1 = (const float*)d_in[5];
    const float* n1b1 = (const float*)d_in[6];
    const float* n1W2 = (const float*)d_in[7];
    const float* n1b2 = (const float*)d_in[8];
    const float* n1W3 = (const float*)d_in[9];
    const float* n1b3 = (const float*)d_in[10];
    const float* n2W0 = (const float*)d_in[11];
    const float* n2b0 = (const float*)d_in[12];
    const float* n2W1 = (const float*)d_in[13];
    const float* n2b1 = (const float*)d_in[14];
    const float* n2W2 = (const float*)d_in[15];
    const float* n2b2 = (const float*)d_in[16];
    const float* n2W3 = (const float*)d_in[17];
    const float* n2b3 = (const float*)d_in[18];
    float* out = (float*)d_out;

    const int blocks = BTOT / NWAVES;   // 4096 blocks * 8 waves = 32768 batch
    tmd_integration_kernel<<<blocks, NTHREADS, 0, stream>>>(
        qT, x1, x2,
        n1W0, n1b0, n1W1, n1b1, n1W2, n1b2, n1W3, n1b3,
        n2W0, n2b0, n2W1, n2b1, n2W2, n2b2, n2W3, n2b3,
        out);
}
